// BiasAttentionModule_57475252355512
// MI455X (gfx1250) — compile-verified, matrix operand order checked by reading, not run on hardware
//
#include <hip/hip_runtime.h>
#include <hip/hip_bf16.h>
#include <math.h>

// ---------------------------------------------------------------------------
// BiasAttention module for MI455X (gfx1250, wave32).
// GEMMs: v_wmma_f32_16x16x32_bf16 (bf16 in-register convert, f32 accum).
// Bias projection over streamed z tensors (537 MB -> the HBM roofline term):
// v_wmma_f32_16x16x4_f32 with 32-deep batched loads for MLP.
// ---------------------------------------------------------------------------

typedef __attribute__((ext_vector_type(16))) __bf16 v16bf;
typedef __attribute__((ext_vector_type(8)))  float  v8f;
typedef __attribute__((ext_vector_type(2)))  float  v2f;

#define NHEADS 12
#define HDIM   32
#define CQ     384
#define NQ     256
#define BB     4
#define ATTN_SCALE 0.17677669529663687f   // 32^-0.5

// ---- fragment loaders ------------------------------------------------------
// 16-bit A/B fragment for 16x16x32: lane&15 = row (M for A, N for B),
// lanes 0-15 hold K={0..7,16..23}, lanes 16-31 hold K={8..15,24..31}.
__device__ __forceinline__ v16bf load_frag_bf16(const float* __restrict__ base,
                                                int ld, int lane) {
  const int r  = lane & 15;
  const int kb = (lane >> 4) * 8;
  const float* __restrict__ row = base + (long)r * ld + kb;
  v16bf f;
#pragma unroll
  for (int i = 0; i < 8; ++i) f[i]     = (__bf16)row[i];
#pragma unroll
  for (int i = 0; i < 8; ++i) f[8 + i] = (__bf16)row[16 + i];
  return f;
}

__device__ __forceinline__ v8f wmma_bf16(v16bf a, v16bf b, v8f c) {
  return __builtin_amdgcn_wmma_f32_16x16x32_bf16(false, a, false, b,
                                                 (short)0, c, false, false);
}

// ---- generic GEMM: C[M,N] = A[M,K] @ W[N,K]^T + bias (+res / gelu) ---------
// grid = (N/64, M/64), block = 128 (4 waves); each wave: 16 rows x 64 cols.
template <int EPI>  // 0: bias, 1: bias+residual, 2: bias+exact GELU
__global__ __launch_bounds__(128) void gemm_kernel(
    const float* __restrict__ A, const float* __restrict__ W,
    const float* __restrict__ bias, const float* __restrict__ R,
    float* __restrict__ C, int M, int N, int K) {
  const int lane  = threadIdx.x & 31;
  const int wave  = threadIdx.x >> 5;
  const int nBase = blockIdx.x * 64;
  const int mBase = blockIdx.y * 64 + wave * 16;

  v8f acc[4] = {};
  const float* __restrict__ Arow = A + (long)mBase * K;
#pragma unroll 2
  for (int k0 = 0; k0 < K; k0 += 32) {
    v16bf a = load_frag_bf16(Arow + k0, K, lane);
#pragma unroll
    for (int j = 0; j < 4; ++j) {
      v16bf b = load_frag_bf16(W + (long)(nBase + j * 16) * K + k0, K, lane);
      acc[j] = wmma_bf16(a, b, acc[j]);
    }
  }
  const int n0   = lane & 15;
  const int mOff = (lane >> 4) * 8;
#pragma unroll
  for (int j = 0; j < 4; ++j) {
    const int n  = nBase + j * 16 + n0;
    const float bv = bias[n];
#pragma unroll
    for (int r = 0; r < 8; ++r) {
      const long idx = (long)(mBase + mOff + r) * N + n;
      float v = acc[j][r] + bv;
      if (EPI == 1) v += R[idx];
      if (EPI == 2) v = 0.5f * v * (1.0f + erff(v * 0.70710678118654752f));
      C[idx] = v;
    }
  }
}

// ---- LayerNorm: one wave per row of 384 ------------------------------------
__global__ __launch_bounds__(256) void ln_kernel(
    const float* __restrict__ X, const float* __restrict__ g,
    const float* __restrict__ b, float* __restrict__ Y, int rows) {
  const int lane = threadIdx.x & 31;
  const long row = (long)blockIdx.x * 8 + (threadIdx.x >> 5);
  if (row >= rows) return;
  const float* __restrict__ x = X + row * CQ;
  float v[12], s = 0.f;
#pragma unroll
  for (int i = 0; i < 12; ++i) { v[i] = x[lane + 32 * i]; s += v[i]; }
#pragma unroll
  for (int o = 16; o; o >>= 1) s += __shfl_xor(s, o, 32);
  const float mu = s * (1.0f / CQ);
  float var = 0.f;
#pragma unroll
  for (int i = 0; i < 12; ++i) { float d = v[i] - mu; var += d * d; }
#pragma unroll
  for (int o = 16; o; o >>= 1) var += __shfl_xor(var, o, 32);
  const float inv = rsqrtf(var * (1.0f / CQ) + 1e-5f);
#pragma unroll
  for (int i = 0; i < 12; ++i) {
    const int c = lane + 32 * i;
    Y[row * CQ + c] = (v[i] - mu) * inv * g[c] + b[c];
  }
}

// ---- V transpose: Vt[(b,h,d),kk] = KV[(b*Nk+kk)*768 + 384 + h*32 + d] ------
__global__ __launch_bounds__(256) void vtrans_kernel(
    const float* __restrict__ KV, float* __restrict__ Vt, int Nk) {
  const long total = (long)BB * NHEADS * HDIM * Nk;
  long i = (long)blockIdx.x * 256 + threadIdx.x;
  if (i >= total) return;
  const int kk = (int)(i % Nk); long t = i / Nk;
  const int d = (int)(t % HDIM); t /= HDIM;
  const int h = (int)(t % NHEADS); const int b = (int)(t / NHEADS);
  Vt[i] = KV[((long)b * Nk + kk) * 768 + 384 + h * 32 + d];
}

// ---- bias projection over z (fp32 WMMA 16x16x4) ----------------------------
// z:(B,Nq,Nk,128) @ W(12,128)^T + b -> biasT:(B,12,Nq,Nk). One wave / 16 rows.
// Two-phase: batch all 32 A-fragment loads (32 outstanding b64 loads, one
// LOADcnt wait), then run the 32-deep WMMA chain -> latency hidden by MLP.
__global__ __launch_bounds__(128) void biasproj_kernel(
    const float* __restrict__ z, const float* __restrict__ W,
    const float* __restrict__ bias, float* __restrict__ biasT,
    int Nq, int Nk) {
  const int lane   = threadIdx.x & 31;
  const long tile  = (long)blockIdx.x * 4 + (threadIdx.x >> 5);
  const long rowBase = tile * 16;                     // flattened (b,q,kk)
  const int  n   = lane & 15;                          // head (pad 12->16)
  const int  kb2 = (lane >> 4) * 2;                    // K offset in 4-window

  v2f bfrag[32];                                       // all 32 K-steps of W^T
#pragma unroll
  for (int k0 = 0; k0 < 32; ++k0) {
#pragma unroll
    for (int i = 0; i < 2; ++i)
      bfrag[k0][i] = (n < NHEADS) ? W[n * 128 + k0 * 4 + kb2 + i] : 0.0f;
  }

  const float* __restrict__ zr = z + (rowBase + (lane & 15)) * 128 + kb2;
  // Phase 1: issue all A-fragment loads (no WMMA in between).
  v2f afrag[32];
#pragma unroll
  for (int k0 = 0; k0 < 32; ++k0) {
    afrag[k0][0] = zr[k0 * 4];
    afrag[k0][1] = zr[k0 * 4 + 1];
  }
  // Phase 2: 32 back-to-back f32 WMMAs.
  v8f acc = {};
#pragma unroll
  for (int k0 = 0; k0 < 32; ++k0) {
    acc = __builtin_amdgcn_wmma_f32_16x16x4_f32(false, afrag[k0], false,
                                                bfrag[k0], (short)0, acc,
                                                false, false);
  }

  const long bq = rowBase / Nk;                        // b*Nq + q
  const int  kkBase = (int)(rowBase % Nk) + (lane >> 4) * 8;
  const int  b = (int)(bq / Nq), q = (int)(bq % Nq);
  if (n < NHEADS) {
    const float bb = bias[n];
    float* __restrict__ dst =
        biasT + (((long)b * NHEADS + n) * Nq + q) * Nk + kkBase;
#pragma unroll
    for (int r = 0; r < 8; ++r) dst[r] = acc[r] + bb;
  }
}

// ---- scores: S[bh,q,kk] = scale * q.k (+biasT), one x32 WMMA per tile ------
// grid = (Nk/64, Nq/64, B*H); wave: 16 q-rows x 64 kk.
__global__ __launch_bounds__(128) void scores_kernel(
    const float* __restrict__ Q, const float* __restrict__ KV,
    const float* __restrict__ biasT, float* __restrict__ S,
    int Nq, int Nk) {
  const int lane  = threadIdx.x & 31;
  const int wave  = threadIdx.x >> 5;
  const int bh = blockIdx.z, b = bh / NHEADS, h = bh % NHEADS;
  const int qBase = blockIdx.y * 64 + wave * 16;
  const int kBase = blockIdx.x * 64;
  const float* __restrict__ Ap = Q  + ((long)b * Nq) * CQ  + h * 32;
  const float* __restrict__ Bp = KV + ((long)b * Nk) * 768 + h * 32;

  v8f acc[4] = {};
  v16bf a = load_frag_bf16(Ap + (long)qBase * CQ, CQ, lane);
#pragma unroll
  for (int j = 0; j < 4; ++j) {
    v16bf bf = load_frag_bf16(Bp + (long)(kBase + j * 16) * 768, 768, lane);
    acc[j] = wmma_bf16(a, bf, acc[j]);
  }
  float* __restrict__ Srow = S + ((long)bh * Nq) * Nk;
  const float* __restrict__ Bi =
      biasT ? biasT + ((long)bh * Nq) * Nk : nullptr;
  const int n0 = lane & 15, mOff = (lane >> 4) * 8;
#pragma unroll
  for (int j = 0; j < 4; ++j) {
#pragma unroll
    for (int r = 0; r < 8; ++r) {
      const int q  = qBase + mOff + r;
      const int kk = kBase + j * 16 + n0;
      float v = acc[j][r] * ATTN_SCALE;
      if (Bi) v += Bi[(long)q * Nk + kk];
      Srow[(long)q * Nk + kk] = v;
    }
  }
}

// ---- softmax in place: one wave per row ------------------------------------
__global__ __launch_bounds__(256) void softmax_kernel(float* __restrict__ S,
                                                      long rows, int Nk) {
  const int lane = threadIdx.x & 31;
  const long row = (long)blockIdx.x * 8 + (threadIdx.x >> 5);
  if (row >= rows) return;
  float* __restrict__ p = S + row * Nk;
  float mx = -1e30f;
  for (int i = lane; i < Nk; i += 32) mx = fmaxf(mx, p[i]);
#pragma unroll
  for (int o = 16; o; o >>= 1) mx = fmaxf(mx, __shfl_xor(mx, o, 32));
  float sum = 0.f;
  for (int i = lane; i < Nk; i += 32) {
    const float e = __expf(p[i] - mx); p[i] = e; sum += e;
  }
#pragma unroll
  for (int o = 16; o; o >>= 1) sum += __shfl_xor(sum, o, 32);
  const float inv = 1.0f / sum;
  for (int i = lane; i < Nk; i += 32) p[i] *= inv;
}

// ---- P @ V: out[b,q,h*32+d] -------------------------------------------------
// grid = (Nq/64, B*H); wave: 16 q-rows x 32 d.
__global__ __launch_bounds__(128) void pv_kernel(
    const float* __restrict__ P, const float* __restrict__ Vt,
    float* __restrict__ O, int Nq, int Nk) {
  const int lane = threadIdx.x & 31;
  const int wave = threadIdx.x >> 5;
  const int bh = blockIdx.y, b = bh / NHEADS, h = bh % NHEADS;
  const int qBase = blockIdx.x * 64 + wave * 16;
  const float* __restrict__ Pp = P  + ((long)bh * Nq) * Nk;
  const float* __restrict__ Vp = Vt + ((long)bh * HDIM) * Nk;

  v8f acc[2] = {};
#pragma unroll 2
  for (int k0 = 0; k0 < Nk; k0 += 32) {
    v16bf a = load_frag_bf16(Pp + (long)qBase * Nk + k0, Nk, lane);
#pragma unroll
    for (int j = 0; j < 2; ++j) {
      v16bf bf = load_frag_bf16(Vp + (long)(j * 16) * Nk + k0, Nk, lane);
      acc[j] = wmma_bf16(a, bf, acc[j]);
    }
  }
  const int n0 = lane & 15, mOff = (lane >> 4) * 8;
#pragma unroll
  for (int j = 0; j < 2; ++j) {
#pragma unroll
    for (int r = 0; r < 8; ++r) {
      const int q = qBase + mOff + r;
      const int d = j * 16 + n0;
      O[((long)b * Nq + q) * CQ + h * 32 + d] = acc[j][r];
    }
  }
}

// ---------------------------------------------------------------------------
// Host orchestration
// ---------------------------------------------------------------------------
struct Blk {
  const float *ln1_g, *ln1_b, *kvw, *kvb, *qw, *qb, *bw, *bb,
              *pw, *pb, *ln2_g, *ln2_b, *f1w, *f1b, *f2w, *f2b;
};

struct Scratch {
  float *ln_buf, *kv_out, *q_out, *v_t, *biasT, *scores, *attn_o, *x_mid, *fc1;
};

static void run_block(const float* xq, const float* xkv, int Nk, int Ckv,
                      const float* z, const Blk& p, float* out,
                      const Scratch& s, hipStream_t stream) {
  const int Mq  = BB * NQ;     // 1024
  const int Mkv = BB * Nk;     // 1024 or 1536
  // 1) LN1 on queries
  ln_kernel<<<dim3(Mq / 8), 256, 0, stream>>>(xq, p.ln1_g, p.ln1_b, s.ln_buf, Mq);
  // 2) kv = xkv @ Wkv^T + b   (raw x_kv per reference)
  gemm_kernel<0><<<dim3(768 / 64, Mkv / 64), 128, 0, stream>>>(
      xkv, p.kvw, p.kvb, nullptr, s.kv_out, Mkv, 768, Ckv);
  // 3) q = ln(xq) @ Wq^T + b
  gemm_kernel<0><<<dim3(CQ / 64, Mq / 64), 128, 0, stream>>>(
      s.ln_buf, p.qw, p.qb, nullptr, s.q_out, Mq, CQ, CQ);
  // 4) V transpose for the PV GEMM
  const long vt_total = (long)BB * NHEADS * HDIM * Nk;
  vtrans_kernel<<<dim3((unsigned)((vt_total + 255) / 256)), 256, 0, stream>>>(
      s.kv_out, s.v_t, Nk);
  // 5) bias projection (streams z once, fused transpose into (B,H,Nq,Nk))
  const float* bptr = nullptr;
  if (z) {
    const long tiles = (long)BB * NQ * (Nk / 16);
    biasproj_kernel<<<dim3((unsigned)(tiles / 4)), 128, 0, stream>>>(
        z, p.bw, p.bb, s.biasT, NQ, Nk);
    bptr = s.biasT;
  }
  // 6) scores = scale*q.k^T + bias
  scores_kernel<<<dim3(Nk / 64, NQ / 64, BB * NHEADS), 128, 0, stream>>>(
      s.q_out, s.kv_out, bptr, s.scores, NQ, Nk);
  // 7) softmax (in place)
  const long srows = (long)BB * NHEADS * NQ;
  softmax_kernel<<<dim3((unsigned)((srows + 7) / 8)), 256, 0, stream>>>(
      s.scores, srows, Nk);
  // 8) out = P @ V
  pv_kernel<<<dim3(NQ / 64, BB * NHEADS), 128, 0, stream>>>(
      s.scores, s.v_t, s.attn_o, NQ, Nk);
  // 9) proj + residual(xq)
  gemm_kernel<1><<<dim3(CQ / 64, Mq / 64), 128, 0, stream>>>(
      s.attn_o, p.pw, p.pb, xq, s.x_mid, Mq, CQ, CQ);
  // 10) LN2
  ln_kernel<<<dim3(Mq / 8), 256, 0, stream>>>(
      s.x_mid, p.ln2_g, p.ln2_b, s.ln_buf, Mq);
  // 11) fc1 + exact GELU
  gemm_kernel<2><<<dim3(1536 / 64, Mq / 64), 128, 0, stream>>>(
      s.ln_buf, p.f1w, p.f1b, nullptr, s.fc1, Mq, 1536, CQ);
  // 12) fc2 + residual(x_mid)
  gemm_kernel<1><<<dim3(CQ / 64, Mq / 64), 128, 0, stream>>>(
      s.fc1, p.f2w, p.f2b, s.x_mid, out, Mq, CQ, 1536);
}

extern "C" void kernel_launch(void* const* d_in, const int* in_sizes, int n_in,
                              void* d_out, int out_size, void* d_ws,
                              size_t ws_size, hipStream_t stream) {
  (void)in_sizes; (void)n_in; (void)out_size; (void)ws_size;
  // Flat input order (setup_inputs dict order, params depth-first):
  // 0:h_feat 1:l_feat 2:z_hh 3:z_hl 4:z_ll 5:z_lh 6:point_feat
  // then 6 blocks x 16 leaves:
  // ln1_g, ln1_b, kv.w, kv.b, q.w, q.b, bias.w, bias.b,
  // proj.w, proj.b, ln2_g, ln2_b, fc1.w, fc1.b, fc2.w, fc2.b
  const float* h_feat = (const float*)d_in[0];
  const float* l_feat = (const float*)d_in[1];
  const float* z_hh   = (const float*)d_in[2];
  const float* z_hl   = (const float*)d_in[3];
  const float* z_ll   = (const float*)d_in[4];
  const float* z_lh   = (const float*)d_in[5];
  const float* point  = (const float*)d_in[6];

  Blk blk[6];
  for (int i = 0; i < 6; ++i) {
    const float** q = (const float**)&blk[i];
    for (int j = 0; j < 16; ++j) q[j] = (const float*)d_in[7 + i * 16 + j];
  }

  // Workspace layout (~58 MB of f32)
  float* w = (float*)d_ws;
  Scratch s;
  s.ln_buf = w; w += (long)1024 * 384;
  s.kv_out = w; w += (long)1536 * 768;
  s.q_out  = w; w += (long)1024 * 384;
  s.v_t    = w; w += (long)BB * NHEADS * HDIM * 384;
  s.biasT  = w; w += (long)BB * NHEADS * NQ * NQ;
  s.scores = w; w += (long)BB * NHEADS * NQ * 384;
  s.attn_o = w; w += (long)1024 * 384;
  s.x_mid  = w; w += (long)1024 * 384;
  s.fc1    = w; w += (long)1024 * 1536;
  float* h1 = w; w += (long)1024 * 384;
  float* l1 = w; w += (long)1024 * 384;
  float* h2 = w; w += (long)1024 * 384;
  float* l2 = w; w += (long)1024 * 384;

  float* out_h = (float*)d_out;                 // (4,256,384)
  float* out_l = out_h + (long)1024 * 384;      // (4,256,384)

  // Reference sequence:
  run_block(h_feat, h_feat, 256, 384, z_hh, blk[0], h1, s, stream);  // h_self
  run_block(l_feat, l_feat, 256, 384, z_ll, blk[1], l1, s, stream);  // l_self
  run_block(h1, point, 384, 128, nullptr, blk[2], h2, s, stream);    // h_point
  run_block(l1, point, 384, 128, nullptr, blk[3], l2, s, stream);    // l_point
  run_block(l2, h2,    256, 384, z_lh,    blk[5], out_l, s, stream); // lh_cross
  run_block(h2, out_l, 256, 384, z_hl,    blk[4], out_h, s, stream); // hl_cross
}
